// A2Module_31361851195652
// MI455X (gfx1250) — matrix operand order repaired and checked by reading, not compiled, not run on hardware
//
#include <hip/hip_runtime.h>

#define Bdim 8
#define Cdim 256
#define Hdim 64
#define Wdim 64
#define NH   8
#define DH   32
#define Lseq 1024
#define NBA  32   // B * A

typedef __attribute__((ext_vector_type(16))) _Float16     v16h;
typedef __attribute__((ext_vector_type(8)))  _Float16     v8h;
typedef __attribute__((ext_vector_type(8)))  float        v8f;
typedef __attribute__((ext_vector_type(4)))  float        v4f;
typedef __attribute__((ext_vector_type(4)))  unsigned int v4u;
typedef __attribute__((ext_vector_type(8)))  unsigned int v8u;

__device__ __forceinline__ v8f vzero8() {
  v8f z;
#pragma unroll
  for (int i = 0; i < 8; ++i) z[i] = 0.0f;
  return z;
}

// A-matrix fragment (16x32 f16, M x K), source row-major [M][stride] in halfs.
// Layout: lane(0-15) row m, K 0..15; lane(16-31) K 16..31 (matches B loaders).
__device__ __forceinline__ v16h load_frag_mk(const _Float16* base, int stride) {
  int lane = threadIdx.x & 31;
  int m = lane & 15, h = lane >> 4;
  const _Float16* r = base + (size_t)m * stride;
  v8h lo = *(const v8h*)(r + 16 * h);
  v8h hi = *(const v8h*)(r + 16 * h + 8);
  v16h f;
#pragma unroll
  for (int i = 0; i < 8; ++i) { f[i] = lo[i]; f[i + 8] = hi[i]; }
  return f;
}

// B-matrix fragment (32x16 f16, K x N), source stored N-major: [N][stride].
__device__ __forceinline__ v16h load_frag_nk(const _Float16* base, int stride) {
  int lane = threadIdx.x & 31;
  int n = lane & 15, h = lane >> 4;
  const _Float16* p = base + (size_t)n * stride + 16 * h;
  v8h lo = *(const v8h*)(p);
  v8h hi = *(const v8h*)(p + 8);
  v16h f;
#pragma unroll
  for (int i = 0; i < 8; ++i) { f[i] = lo[i]; f[i + 8] = hi[i]; }
  return f;
}

// Same as load_frag_nk but from an f32 tile (TDM-staged), convert on read.
__device__ __forceinline__ v16h load_frag_nk_f32(const float* base, int stride) {
  int lane = threadIdx.x & 31;
  int n = lane & 15, h = lane >> 4;
  const float* p = base + (size_t)n * stride + 16 * h;
  v16h f;
#pragma unroll
  for (int i = 0; i < 16; ++i) f[i] = (_Float16)p[i];
  return f;
}

__device__ __forceinline__ v8f wmma_f16(v16h a, v16h b, v8f c) {
  return __builtin_amdgcn_wmma_f32_16x16x32_f16(false, a, false, b, (short)0, c,
                                                false, false);
}

// ---------------------------------------------------------------------------
// Kernel 1: per-area QKV in-projection.  qkv[l,d] = xa[l,:] . W_in[a,d,:] + b
// Tile: 64 (l) x 128 (d), K stepped by 32 through LDS (f32 -> f16 convert).
// ---------------------------------------------------------------------------
__global__ __launch_bounds__(256) void qkv_kernel(
    const float* __restrict__ x, const float* __restrict__ Win,
    const float* __restrict__ bin, _Float16* __restrict__ qws,
    _Float16* __restrict__ kws, _Float16* __restrict__ vws) {
  __shared__ __align__(16) _Float16 Ax[64][40];
  __shared__ __align__(16) _Float16 Bw[128][40];

  const int dt = blockIdx.x;   // 0..5   (128-wide d tile)
  const int lt = blockIdx.y;   // 0..15  (64-row l tile)
  const int ba = blockIdx.z;   // 0..31
  const int b = ba >> 2, a = ba & 3;
  const int ai = a >> 1, aj = a & 1;
  const int t = threadIdx.x, wave = t >> 5, lane = t & 31, hlf = lane >> 4;
  const int mt = wave >> 1;            // 0..3 -> 16-row subtile
  const int nb = (wave & 1) * 64;      // n base within 128

  // staging ownership (fixed across K loop)
  const int sc  = t >> 3;          // A tile: channel 0..31
  const int sl0 = (t & 7) * 8;     // A tile: 8 consecutive l rows (aligned)
  const int sp  = (lt * 64 + sl0) >> 5;
  const int sq0 = (lt * 64 + sl0) & 31;
  const int sd  = t >> 1;          // B tile: weight row 0..127
  const int scb = (t & 1) * 16;    // B tile: 16 consecutive c

  v8f acc[4];
#pragma unroll
  for (int i = 0; i < 4; ++i) acc[i] = vzero8();

  for (int kk = 0; kk < Cdim; kk += 32) {
    // --- A tile: two float4 loads (32B-aligned, coalesced in q) ---
    {
      const float* src = &x[(((size_t)b * Cdim + (kk + sc)) * Hdim + ai * 32 +
                             sp) * Wdim + aj * 32 + sq0];
      v4f f0 = *(const v4f*)(src);
      v4f f1 = *(const v4f*)(src + 4);
#pragma unroll
      for (int i = 0; i < 4; ++i) {
        Ax[sl0 + i][sc]     = (_Float16)f0[i];
        Ax[sl0 + 4 + i][sc] = (_Float16)f1[i];
      }
    }
    // --- B tile: 16 consecutive f32 -> 16 halfs, two b128 DS stores ---
    {
      const float* src =
          &Win[((size_t)a * 768 + dt * 128 + sd) * Cdim + kk + scb];
      v4f f0 = *(const v4f*)(src);
      v4f f1 = *(const v4f*)(src + 4);
      v4f f2 = *(const v4f*)(src + 8);
      v4f f3 = *(const v4f*)(src + 12);
      v8h h0, h1;
#pragma unroll
      for (int i = 0; i < 4; ++i) {
        h0[i] = (_Float16)f0[i]; h0[4 + i] = (_Float16)f1[i];
        h1[i] = (_Float16)f2[i]; h1[4 + i] = (_Float16)f3[i];
      }
      *(v8h*)&Bw[sd][scb]     = h0;
      *(v8h*)&Bw[sd][scb + 8] = h1;
    }
    if (kk + 32 < Cdim)
      __builtin_prefetch(&Win[((size_t)a * 768 + dt * 128 + sd) * Cdim + kk +
                              32 + scb], 0, 1);
    __syncthreads();

    v16h af = load_frag_mk(&Ax[mt * 16][0], 40);
#pragma unroll
    for (int nt = 0; nt < 4; ++nt) {
      v16h bf = load_frag_nk(&Bw[nb + nt * 16][0], 40);
      acc[nt] = wmma_f16(af, bf, acc[nt]);
    }
    __syncthreads();
  }

  // bias, split q/k/v, head-major store (q pre-scaled by 1/sqrt(dh))
#pragma unroll
  for (int nt = 0; nt < 4; ++nt)
#pragma unroll
    for (int r = 0; r < 8; ++r) {
      int dg = dt * 128 + nb + nt * 16 + (lane & 15);
      int lg = lt * 64 + mt * 16 + hlf * 8 + r;
      float v = acc[nt][r] + bin[a * 768 + dg];
      int which = dg >> 8;   // 0=q 1=k 2=v
      int cl = dg & 255;
      int hd = cl >> 5, dd = cl & 31;
      if (which == 0) v *= 0.17677669529663687f;  // 1/sqrt(32)
      _Float16* dst = (which == 0) ? qws : ((which == 1) ? kws : vws);
      dst[(((size_t)ba * NH + hd) * Lseq + lg) * DH + dd] = (_Float16)v;
    }
}

// ---------------------------------------------------------------------------
// Kernel 2: flash attention per (b,a,head, 128-query block).
// K tile staged with async global->LDS copies (ASYNCcnt path).
// ---------------------------------------------------------------------------
__global__ __launch_bounds__(256) void attn_kernel(
    const _Float16* __restrict__ qws, const _Float16* __restrict__ kws,
    const _Float16* __restrict__ vws, _Float16* __restrict__ ows) {
  __shared__ __align__(16) _Float16 Kt[128][40];      // [key][dh]  == [N][K]
  __shared__ __align__(16) _Float16 Vt[32][136];      // [dh][key]  == [N][K]
  __shared__ __align__(16) _Float16 Pl[8][16][136];   // per-wave P stage

  const int lb = blockIdx.x;    // query block 0..7
  const int head = blockIdx.y;  // 0..7
  const int ba = blockIdx.z;    // 0..31
  const int t = threadIdx.x, wave = t >> 5, lane = t & 31, hlf = lane >> 4;

  const size_t hb = ((size_t)ba * NH + head) * (size_t)Lseq * DH;
  const _Float16* qb = qws + hb;
  const _Float16* kb = kws + hb;
  const _Float16* vb = vws + hb;

  const unsigned kt_lds = (unsigned)(uintptr_t)(&Kt[0][0]);

  // persistent Q fragment: rows lb*128 + wave*16 .. +15
  v16h qf = load_frag_mk(qb + (size_t)(lb * 128 + wave * 16) * DH, DH);

  float rmax[8], rsum[8];
#pragma unroll
  for (int r = 0; r < 8; ++r) { rmax[r] = -1e30f; rsum[r] = 0.0f; }
  v8f acc[2];
  acc[0] = vzero8();
  acc[1] = vzero8();

  for (int kblk = 0; kblk < 8; ++kblk) {
    // --- K tile via async global->LDS B128 copies (16B per chunk) ---
#pragma unroll
    for (int i = 0; i < 2; ++i) {
      int e = t * 2 + i;                 // 0..511 chunks
      int row = e >> 2, part = e & 3;    // 4 x 16B per 32-half row
      const _Float16* g = kb + (size_t)(kblk * 128 + row) * DH + part * 8;
      unsigned lds = kt_lds + row * 80 + part * 16;  // padded row = 80B
      asm volatile("global_load_async_to_lds_b128 %0, %1, off"
                   :: "v"(lds), "v"(g) : "memory");
    }
    // --- V tile transposed: vectorized read, scatter DS stores ---
    {
      int key = t >> 1, db = (t & 1) * 16;
      const _Float16* src = vb + (size_t)(kblk * 128 + key) * DH + db;
      v8h a0 = *(const v8h*)(src);
      v8h a1 = *(const v8h*)(src + 8);
#pragma unroll
      for (int i = 0; i < 8; ++i) {
        Vt[db + i][key]     = a0[i];
        Vt[db + 8 + i][key] = a1[i];
      }
    }
    asm volatile("s_wait_asynccnt 0x0" ::: "memory");
    __syncthreads();

    // scores: 8 tiles of 16 keys, single K=32 WMMA each
    v8f s[8];
#pragma unroll
    for (int nt = 0; nt < 8; ++nt) {
      v16h kf = load_frag_nk(&Kt[nt * 16][0], 40);
      s[nt] = wmma_f16(qf, kf, vzero8());
    }

    // online softmax per row (row r+8*hlf lives across 16 lanes of this half)
#pragma unroll
    for (int r = 0; r < 8; ++r) {
      float mx = -1e30f;
#pragma unroll
      for (int nt = 0; nt < 8; ++nt) mx = fmaxf(mx, s[nt][r]);
      mx = fmaxf(mx, __shfl_xor(mx, 1, 32));
      mx = fmaxf(mx, __shfl_xor(mx, 2, 32));
      mx = fmaxf(mx, __shfl_xor(mx, 4, 32));
      mx = fmaxf(mx, __shfl_xor(mx, 8, 32));
      float nm = fmaxf(rmax[r], mx);
      float alpha = __expf(rmax[r] - nm);
      rmax[r] = nm;
      float ps = 0.0f;
#pragma unroll
      for (int nt = 0; nt < 8; ++nt) {
        float pv = __expf(s[nt][r] - nm);
        s[nt][r] = pv;
        ps += pv;
      }
      ps += __shfl_xor(ps, 1, 32);
      ps += __shfl_xor(ps, 2, 32);
      ps += __shfl_xor(ps, 4, 32);
      ps += __shfl_xor(ps, 8, 32);
      rsum[r] = rsum[r] * alpha + ps;
      acc[0][r] *= alpha;
      acc[1][r] *= alpha;
    }

    // D-layout -> A-layout via per-wave LDS stage (f16)
#pragma unroll
    for (int nt = 0; nt < 8; ++nt)
#pragma unroll
      for (int r = 0; r < 8; ++r)
        Pl[wave][hlf * 8 + r][nt * 16 + (lane & 15)] = (_Float16)s[nt][r];
    __syncthreads();

    // O += P (16x128) . V (128x32): 2 N-tiles x 4 K-chunks
#pragma unroll
    for (int ot = 0; ot < 2; ++ot)
#pragma unroll
      for (int kc = 0; kc < 4; ++kc) {
        v16h pf = load_frag_mk(&Pl[wave][0][kc * 32], 136);
        v16h vf = load_frag_nk(&Vt[ot * 16][kc * 32], 136);
        acc[ot] = wmma_f16(pf, vf, acc[ot]);
      }
    __syncthreads();
  }

  // normalize and write O as f16 into [ba][L][C], channel = head*32 + n
#pragma unroll
  for (int ot = 0; ot < 2; ++ot)
#pragma unroll
    for (int r = 0; r < 8; ++r) {
      int row = lb * 128 + wave * 16 + hlf * 8 + r;
      int ch = head * DH + ot * 16 + (lane & 15);
      ows[((size_t)ba * Lseq + row) * Cdim + ch] =
          (_Float16)(acc[ot][r] / rsum[r]);
    }
}

// ---------------------------------------------------------------------------
// Kernel 3: out-projection + residual + channel LayerNorm + area reassembly.
// W_out tile (256x32 f32) staged each K-step by the Tensor Data Mover with
// 4-DWORD row padding (36-DWORD rows); LayerNorm tile aliases the same LDS.
// ---------------------------------------------------------------------------
__global__ __launch_bounds__(256) void out_kernel(
    const _Float16* __restrict__ ows, const float* __restrict__ x,
    const float* __restrict__ Wout, const float* __restrict__ bout,
    const float* __restrict__ gamma, const float* __restrict__ beta,
    float* __restrict__ out) {
  __shared__ __align__(16) char smem[256 * 36 * 4];  // 36864 B, two phases
  float* Bwf = (float*)smem;                         // [256][36] f32 tile

  const int lblk = blockIdx.x;  // 0..31 (32 rows each)
  const int ba = blockIdx.y;    // 0..31
  const int b = ba >> 2, a = ba & 3, ai = a >> 1, aj = a & 1;
  const int t = threadIdx.x, wave = t >> 5, lane = t & 31, hlf = lane >> 4;
  const int mt = wave >> 2;          // 0..1
  const int nb = (wave & 3) * 64;    // n base within 256

  v8f acc[4];
#pragma unroll
  for (int i = 0; i < 4; ++i) acc[i] = vzero8();

  const _Float16* ob = ows + ((size_t)ba * Lseq + lblk * 32 + mt * 16) * Cdim;
  const unsigned lds_b = (unsigned)(uintptr_t)smem;

  for (int kk = 0; kk < Cdim; kk += 32) {
    __syncthreads();  // previous tile fully consumed before TDM overwrites
    if (wave == 0) {
      unsigned long long ga =
          (unsigned long long)(uintptr_t)(Wout + (size_t)a * Cdim * Cdim + kk);
      v4u g0;
      g0[0] = 1u;                         // count=1, user descriptor
      g0[1] = lds_b;                      // lds_addr
      g0[2] = (unsigned)ga;               // global_addr[31:0]
      g0[3] = (unsigned)((ga >> 32) & 0x1ffffffull) | (2u << 30);  // type=2
      v8u g1;
      g1[0] = (2u << 16)        // data_size = 4B
            | (1u << 20)        // pad_enable
            | (4u << 22)        // pad_interval: 32 DWORDs (one 128B row)
            | (3u << 25);       // pad_amount: 4 DWORDs -> 36-DWORD rows
      g1[1] = 32u << 16;        // tensor_dim0 = 32 (tile-sized)
      g1[2] = 256u << 16;       // tensor_dim1 = 256
      g1[3] = 32u << 16;        // tile_dim0 = 32
      g1[4] = 256u;             // tile_dim1 = 256 (tile_dim2 = 0)
      g1[5] = 256u;             // tensor_dim0_stride = 256 elements
      g1[6] = 0u;
      g1[7] = 0u;
      asm volatile("tensor_load_to_lds %0, %1" :: "s"(g0), "s"(g1) : "memory");
    }
    __builtin_amdgcn_s_wait_tensorcnt(0);
    __syncthreads();

    v16h af = load_frag_mk(ob + kk, Cdim);  // A direct from global f16
#pragma unroll
    for (int nt = 0; nt < 4; ++nt) {
      v16h bf = load_frag_nk_f32(Bwf + (size_t)(nb + nt * 16) * 36, 36);
      acc[nt] = wmma_f16(af, bf, acc[nt]);
    }
  }
  __syncthreads();  // done with Bwf; smem becomes the LayerNorm tile

  float (*Y)[260] = (float(*)[260])smem;  // 32*260*4 = 33280 B <= 36864 B

  // bias + residual into LDS y-tile
#pragma unroll
  for (int nt = 0; nt < 4; ++nt)
#pragma unroll
    for (int r = 0; r < 8; ++r) {
      int ch = nb + nt * 16 + (lane & 15);
      int rl = mt * 16 + hlf * 8 + r;
      int lg = lblk * 32 + rl;
      int p = lg >> 5, q = lg & 31;
      float res = x[(((size_t)b * Cdim + ch) * Hdim + ai * 32 + p) * Wdim +
                    aj * 32 + q];
      Y[rl][ch] = acc[nt][r] + bout[a * Cdim + ch] + res;
    }
  __syncthreads();

  // LayerNorm over 256 channels: 8 threads per row, shuffle-reduce
  {
    int r = t >> 3, seg = t & 7;
    float sm = 0.0f, sq = 0.0f;
#pragma unroll
    for (int c0 = 0; c0 < 32; ++c0) {
      float v = Y[r][seg * 32 + c0];
      sm += v;
      sq += v * v;
    }
    sm += __shfl_xor(sm, 1, 32); sq += __shfl_xor(sq, 1, 32);
    sm += __shfl_xor(sm, 2, 32); sq += __shfl_xor(sq, 2, 32);
    sm += __shfl_xor(sm, 4, 32); sq += __shfl_xor(sq, 4, 32);
    float mu = sm * (1.0f / 256.0f);
    float var = sq * (1.0f / 256.0f) - mu * mu;
    float rstd = rsqrtf(var + 1e-5f);
    int lg = lblk * 32 + r;
    int p = lg >> 5, q = lg & 31;
#pragma unroll
    for (int c0 = 0; c0 < 32; ++c0) {
      int ch = seg * 32 + c0;
      float v = (Y[r][ch] - mu) * rstd * gamma[ch] + beta[ch];
      out[(((size_t)b * Cdim + ch) * Hdim + ai * 32 + p) * Wdim + aj * 32 + q] =
          v;
    }
  }
}

// ---------------------------------------------------------------------------
extern "C" void kernel_launch(void* const* d_in, const int* in_sizes, int n_in,
                              void* d_out, int out_size, void* d_ws,
                              size_t ws_size, hipStream_t stream) {
  const float* x     = (const float*)d_in[0];
  const float* Win   = (const float*)d_in[1];
  const float* bin   = (const float*)d_in[2];
  const float* Wout  = (const float*)d_in[3];
  const float* bout  = (const float*)d_in[4];
  const float* gamma = (const float*)d_in[5];
  const float* beta  = (const float*)d_in[6];
  float* out = (float*)d_out;

  // workspace: q, k, v (f16 head-major) + o (f16, [ba][L][C]) = ~67 MB
  const size_t perT = (size_t)NBA * NH * Lseq * DH;  // 8,388,608 halfs each
  _Float16* qws = (_Float16*)d_ws;
  _Float16* kws = qws + perT;
  _Float16* vws = kws + perT;
  _Float16* ows = vws + perT;

  qkv_kernel<<<dim3(6, 16, 32), 256, 0, stream>>>(x, Win, bin, qws, kws, vws);
  attn_kernel<<<dim3(8, NH, 32), 256, 0, stream>>>(qws, kws, vws, ows);
  out_kernel<<<dim3(32, 32), 256, 0, stream>>>(ows, x, Wout, bout, gamma, beta,
                                               out);
}
